// Adaptive_weight_block_24558622998624
// MI455X (gfx1250) — matrix operand order repaired and checked by reading, compiled, not run on hardware
//
#include <hip/hip_runtime.h>
#include <hip/hip_bf16.h>

typedef __attribute__((ext_vector_type(16))) _Float16 v16h;
typedef __attribute__((ext_vector_type(8)))  _Float16 v8h;
typedef __attribute__((ext_vector_type(8)))  float    v8f;

#define LDA 264   // f16 row stride for K=256 tiles (132 DWORDs -> conflict-free frags)
#define LDH 72    // f16 row stride for K=64 tiles
#define LDMP 68   // f32 stride, mixing-policy buffer
#define LDL 52    // f32 stride, logits buffer

// ---- WMMA fragment helpers (layouts per CDNA5 ISA 7.12.2, wave32) ----

// A fragment 16x32 f16 from row-major [row][K], 16B-aligned rows.
__device__ inline v16h load_frag_a(const _Float16* base, int lda, int k0, int lane) {
    const _Float16* p = base + (lane & 15) * lda + k0 + ((lane >> 4) << 3);
    v8h lo = *(const v8h*)(p);
    v8h hi = *(const v8h*)(p + 16);
    return __builtin_shufflevector(lo, hi, 0,1,2,3,4,5,6,7,8,9,10,11,12,13,14,15);
}

// B fragment 32x16 f16 from N-major (transposed) [nCol][K] storage.
__device__ inline v16h load_frag_bt(const _Float16* base, int ldb, int k0, int lane) {
    const _Float16* p = base + (lane & 15) * ldb + k0 + ((lane >> 4) << 3);
    v8h lo = *(const v8h*)(p);
    v8h hi = *(const v8h*)(p + 16);
    return __builtin_shufflevector(lo, hi, 0,1,2,3,4,5,6,7,8,9,10,11,12,13,14,15);
}

__device__ inline v8f wmma_f16(v16h a, v16h b, v8f c) {
    return __builtin_amdgcn_wmma_f32_16x16x32_f16(false, a, false, b, (short)0, c, false, false);
}

__device__ inline float gelu_exact(float x) {
    return 0.5f * x * (1.0f + erff(x * 0.70710678118654752f));
}

// cooperative stage: global f16 [rows x kHalves] (row pitch srcPitch) -> LDS [rows x dstStride]
__device__ inline void stage_bh(const _Float16* __restrict__ g, int srcPitch,
                                int rows, int kHalves, int dstStride,
                                _Float16* __restrict__ dst, int tid) {
    const int cpr = kHalves >> 3;           // 16B chunks per row
    const int total = rows * cpr;
    for (int c = tid; c < total; c += 256) {
        const int r = c / cpr;
        const int o = (c - r * cpr) << 3;
        *(v8h*)(dst + r * dstStride + o) = *(const v8h*)(g + (size_t)r * srcPitch + o);
    }
}

// ---------------- prep: transpose weights to f16 N-major + gather bank --------
__global__ __launch_bounds__(256) void prep_kernel(
    const float* __restrict__ w1,   const float* __restrict__ w2,
    const float* __restrict__ prew, const float* __restrict__ postw,
    const float* __restrict__ fc1w, const float* __restrict__ fc2w,
    const float* __restrict__ wbank,const int*   __restrict__ relidx,
    _Float16* __restrict__ w1t, _Float16* __restrict__ w2t,
    _Float16* __restrict__ pret, _Float16* __restrict__ postt,
    _Float16* __restrict__ fc1t, _Float16* __restrict__ fc2t,
    float* __restrict__ wbg)
{
    int i = blockIdx.x * 256 + threadIdx.x;
    if (i < 16384) { int n = i >> 8, k = i & 255; w1t[i]  = (_Float16)w1[k * 64 + n];    return; }
    i -= 16384;
    if (i < 4096)  { int n = i >> 6, k = i & 63;  w2t[i]  = (_Float16)w2[k * 64 + n];    return; }
    i -= 4096;
    if (i < 65536) { int n = i >> 8, k = i & 255; pret[i] = (_Float16)prew[k * 256 + n]; return; }
    i -= 65536;
    if (i < 65536) { int n = i >> 8, k = i & 255; postt[i]= (_Float16)postw[k * 256 + n];return; }
    i -= 65536;
    if (i < 262144){ int n = i >> 8, k = i & 255; fc1t[i] = (_Float16)fc1w[k * 1024 + n];return; }
    i -= 262144;
    if (i < 262144){ int n = i >> 10,k = i & 1023;fc2t[i] = (_Float16)fc2w[k * 256 + n]; return; }
    i -= 262144;
    if (i < 19208) { int k = i / 2401, r = i % 2401, n = r / 49, m = r % 49;
                     wbg[i] = wbank[k * 169 + relidx[n * 49 + m]]; }
}

// ---------------- kernel 1: fused LN1 + adapter + adaptive mixing + residual --
// LDS map (bytes):
//   0        sXn   64x264 f16  (33,792)
//   33792    sXpT 256x72  f16  (36,864)
//   70656    sOut  64x264 f16  (33,792)
//   104448   sH    64x72  f16  ( 9,216)
//   113664   sMp   64x68  f32  (17,408)
//   131072   sL    64x52  f32  (13,312)
//   144384   sWt   64x72  f16  ( 9,216)
//   153600   sB   256x264 f16  (135,168)  staged weights / gathered bank
//   total 288,768
__global__ __launch_bounds__(256) void win_kernel(
    const float* __restrict__ x,
    const float* __restrict__ n1s, const float* __restrict__ n1b,
    const float* __restrict__ gamma1,
    const float* __restrict__ b1,  const float* __restrict__ b2,
    const float* __restrict__ pre_b, const float* __restrict__ post_b,
    const _Float16* __restrict__ w1t, const _Float16* __restrict__ w2t,
    const _Float16* __restrict__ pre_wt, const _Float16* __restrict__ post_wt,
    const float* __restrict__ wbg,
    float* __restrict__ x1)
{
    extern __shared__ char smem[];
    _Float16* sXn  = (_Float16*)(smem);
    _Float16* sXpT = (_Float16*)(smem + 33792);
    _Float16* sOut = (_Float16*)(smem + 70656);
    _Float16* sH   = (_Float16*)(smem + 104448);
    float*    sMp  = (float*)   (smem + 113664);
    float*    sL   = (float*)   (smem + 131072);
    _Float16* sWt  = (_Float16*)(smem + 144384);
    _Float16* sB   = (_Float16*)(smem + 153600);
    float*    sBf  = (float*)   (smem + 153600);

    const int tid  = threadIdx.x;
    const int lane = tid & 31;
    const int wave = tid >> 5;
    const int nl   = lane & 15;
    const int mo   = (lane >> 4) * 8;

    const int win = blockIdx.x;
    const int b   = win >> 6;
    const int wr  = (win >> 3) & 7;
    const int wc  = win & 7;
    const int pr0 = wr * 7, pc0 = wc * 7;

    // zero pad rows 49..63 of sXn
    for (int i = tid; i < 15 * LDA; i += 256) sXn[49 * LDA + i] = (_Float16)0.0f;

    // ---- LN1 -> f16 (one token per wave, shfl-xor reduction) ----
    for (int t = wave; t < 49; t += 8) {
        const int gr = pr0 + t / 7, gc = pc0 + t % 7;
        const float* px = x + (((size_t)b * 56 + gr) * 56 + gc) * 256;
        float v[8], s = 0.f, s2 = 0.f;
#pragma unroll
        for (int i = 0; i < 8; ++i) { v[i] = px[lane * 8 + i]; s += v[i]; s2 += v[i] * v[i]; }
#pragma unroll
        for (int m = 16; m >= 1; m >>= 1) { s += __shfl_xor(s, m); s2 += __shfl_xor(s2, m); }
        const float mu  = s * (1.f / 256.f);
        const float var = s2 * (1.f / 256.f) - mu * mu;
        const float rs  = rsqrtf(var + 1e-5f);
#pragma unroll
        for (int i = 0; i < 8; ++i) {
            const int c = lane * 8 + i;
            sXn[t * LDA + c] = (_Float16)((v[i] - mu) * rs * n1s[c] + n1b[c]);
        }
    }
    stage_bh(w1t, 256, 64, 256, LDA, sB, tid);     // stage adapter W1 (N-major)
    __syncthreads();

    // ---- adapter GEMM1 (k-outer, hoisted A): gelu(xn @ w1 + b1) -> sH ----
    {
        const int mt = wave >> 1, nt0 = (wave & 1) * 2;
        v8f hc[2] = {};
        for (int k = 0; k < 8; ++k) {
            v16h a = load_frag_a(sXn + mt * 16 * LDA, LDA, k * 32, lane);
#pragma unroll
            for (int i = 0; i < 2; ++i) {
                v16h bb = load_frag_bt(sB + (nt0 + i) * 16 * LDA, LDA, k * 32, lane);
                hc[i] = wmma_f16(a, bb, hc[i]);
            }
        }
#pragma unroll
        for (int i = 0; i < 2; ++i) {
            const int col = (nt0 + i) * 16 + nl;
            const float bias = b1[col];
#pragma unroll
            for (int r = 0; r < 8; ++r)
                sH[(mt * 16 + mo + r) * LDH + col] = (_Float16)gelu_exact(hc[i][r] + bias);
        }
    }
    __syncthreads();
    stage_bh(w2t, 64, 64, 64, LDH, sB, tid);       // stage adapter W2
    __syncthreads();

    // ---- adapter GEMM2: sH @ w2 + b2 -> sMp (f32) ----
    {
        const int mt = wave >> 1, nt0 = (wave & 1) * 2;
        v8f hc[2] = {};
        for (int k = 0; k < 2; ++k) {
            v16h a = load_frag_a(sH + mt * 16 * LDH, LDH, k * 32, lane);
#pragma unroll
            for (int i = 0; i < 2; ++i) {
                v16h bb = load_frag_bt(sB + (nt0 + i) * 16 * LDH, LDH, k * 32, lane);
                hc[i] = wmma_f16(a, bb, hc[i]);
            }
        }
#pragma unroll
        for (int i = 0; i < 2; ++i) {
            const int col = (nt0 + i) * 16 + nl;
            const float bias = b2[col];
#pragma unroll
            for (int r = 0; r < 8; ++r) sMp[(mt * 16 + mo + r) * LDMP + col] = hc[i][r] + bias;
        }
    }
    __syncthreads();
    stage_bh(pre_wt, 256, 256, 256, LDA, sB, tid); // stage pre-proj weights
    __syncthreads();

    // ---- pre-proj GEMM (k-outer, hoisted A): xn @ pre_w + pre_b -> sXpT (transposed)
    {
        const int mt = wave >> 1, nt0 = (wave & 1) * 8;
        v8f hc[8] = {};
        for (int k = 0; k < 8; ++k) {
            v16h a = load_frag_a(sXn + mt * 16 * LDA, LDA, k * 32, lane);
#pragma unroll
            for (int i = 0; i < 8; ++i) {
                v16h bb = load_frag_bt(sB + (nt0 + i) * 16 * LDA, LDA, k * 32, lane);
                hc[i] = wmma_f16(a, bb, hc[i]);
            }
        }
#pragma unroll
        for (int i = 0; i < 8; ++i) {
            const int col = (nt0 + i) * 16 + nl;
            const float bias = pre_b[col];
            _Float16* dst = sXpT + col * LDH + mt * 16 + mo;  // [chan][token]
#pragma unroll
            for (int r = 0; r < 8; ++r) dst[r] = (_Float16)(hc[i][r] + bias);
        }
    }
    __syncthreads();
    // stage gathered relative-position bank (f32, 19208 elems) into sB region
    {
        const float4* src = (const float4*)wbg;
        float4* dv = (float4*)sBf;
        for (int c = tid; c < 4802; c += 256) dv[c] = src[c];
    }
    __builtin_prefetch(post_wt + (size_t)tid * 256, 0, 1);  // warm GL2 for post weights
    __syncthreads();

    // ---- per-head: logits (VALU, all-LDS) -> softmax over n -> WMMA mixing ----
    for (int h = 0; h < 8; ++h) {
        for (int idx = tid; idx < 2401; idx += 256) {
            const int m = idx / 49, n = idx - m * 49;
            float s = 0.f;
#pragma unroll
            for (int k = 0; k < 8; ++k)
                s += sMp[n * LDMP + k * 8 + h] * sBf[(k * 49 + n) * 49 + m];
            sL[m * LDL + n] = s;
        }
        __syncthreads();
        if (tid < 64) {
            const int m = tid;
            if (m < 49) {
                float mx = -1e30f;
                for (int n = 0; n < 49; ++n) mx = fmaxf(mx, sL[m * LDL + n]);
                float ssum = 0.f;
                for (int n = 0; n < 49; ++n) ssum += __expf(sL[m * LDL + n] - mx);
                const float inv = 1.f / ssum;
                for (int n = 0; n < 64; ++n)
                    sWt[m * LDH + n] = (n < 49) ? (_Float16)(__expf(sL[m * LDL + n] - mx) * inv)
                                                : (_Float16)0.0f;
            } else {
                for (int n = 0; n < 64; ++n) sWt[m * LDH + n] = (_Float16)0.0f;
            }
        }
        __syncthreads();
        {   // out_h(64x32) = W^T(64x64) @ xp_h(64x32) ; one tile per wave
            const int mt = wave >> 1, nt = wave & 1;
            v8f acc = {};
#pragma unroll
            for (int k = 0; k < 2; ++k) {
                v16h a  = load_frag_a(sWt + mt * 16 * LDH, LDH, k * 32, lane);
                v16h bb = load_frag_bt(sXpT + (h * 32 + nt * 16) * LDH, LDH, k * 32, lane);
                acc = wmma_f16(a, bb, acc);
            }
            const int col = h * 32 + nt * 16 + nl;
#pragma unroll
            for (int r = 0; r < 8; ++r)
                sOut[(mt * 16 + mo + r) * LDA + col] = (_Float16)acc[r];
        }
        __syncthreads();
    }

    stage_bh(post_wt, 256, 256, 256, LDA, sB, tid); // stage post-proj weights
    __syncthreads();

    // ---- post-proj GEMM + residual: x1 = x + gamma1*(out@post_w + post_b) ----
    {
        const int mt = wave >> 1, nt0 = (wave & 1) * 8;
        v8f hc[8] = {};
        for (int k = 0; k < 8; ++k) {
            v16h a = load_frag_a(sOut + mt * 16 * LDA, LDA, k * 32, lane);
#pragma unroll
            for (int i = 0; i < 8; ++i) {
                v16h bb = load_frag_bt(sB + (nt0 + i) * 16 * LDA, LDA, k * 32, lane);
                hc[i] = wmma_f16(a, bb, hc[i]);
            }
        }
#pragma unroll
        for (int i = 0; i < 8; ++i) {
            const int col = (nt0 + i) * 16 + nl;
            const float bias = post_b[col], g = gamma1[col];
#pragma unroll
            for (int r = 0; r < 8; ++r) {
                const int m = mt * 16 + mo + r;
                if (m < 49) {
                    const int gr = pr0 + m / 7, gc = pc0 + m % 7;
                    const size_t gi = (((size_t)b * 56 + gr) * 56 + gc) * 256 + col;
                    x1[gi] = x[gi] + g * (hc[i][r] + bias);
                }
            }
        }
    }
}

// ---------------- kernel 2: fused LN2 + MLP(256->1024->256) + residual --------
// 128 tokens per block; weights staged chunk-wise into LDS (read exactly once).
// LDS: sA 128x264 (67,584) | sH 128x264 (67,584) | sB 256x264 (135,168) = 270,336
__global__ __launch_bounds__(256) void mlp_kernel(
    const float* __restrict__ x1,
    const float* __restrict__ n2s, const float* __restrict__ n2b,
    const float* __restrict__ gamma2,
    const float* __restrict__ fc1b, const float* __restrict__ fc2b,
    const _Float16* __restrict__ fc1t, const _Float16* __restrict__ fc2t,
    float* __restrict__ out)
{
    extern __shared__ char smem[];
    _Float16* sA = (_Float16*)(smem);
    _Float16* sH = (_Float16*)(smem + 67584);
    _Float16* sB = (_Float16*)(smem + 135168);

    const int tid  = threadIdx.x;
    const int lane = tid & 31;
    const int wave = tid >> 5;
    const int nl   = lane & 15;
    const int mo   = (lane >> 4) * 8;
    const size_t T0 = (size_t)blockIdx.x * 128;
    const int mt = wave;                 // wave owns row-tile stripe (16 rows)

    // ---- LN2 -> f16 ----
    for (int t = wave; t < 128; t += 8) {
        const float* px = x1 + (T0 + t) * 256;
        float v[8], s = 0.f, s2 = 0.f;
#pragma unroll
        for (int i = 0; i < 8; ++i) { v[i] = px[lane * 8 + i]; s += v[i]; s2 += v[i] * v[i]; }
#pragma unroll
        for (int m = 16; m >= 1; m >>= 1) { s += __shfl_xor(s, m); s2 += __shfl_xor(s2, m); }
        const float mu  = s * (1.f / 256.f);
        const float var = s2 * (1.f / 256.f) - mu * mu;
        const float rs  = rsqrtf(var + 1e-5f);
#pragma unroll
        for (int i = 0; i < 8; ++i) {
            const int c = lane * 8 + i;
            sA[t * LDA + c] = (_Float16)((v[i] - mu) * rs * n2s[c] + n2b[c]);
        }
    }

    v8f acc[16] = {};                    // persistent 128x256 f32 output accumulators

    for (int c4 = 0; c4 < 4; ++c4) {     // hid=1024 in 4 chunks of 256
        stage_bh(fc1t + (size_t)(c4 * 256) * 256, 256, 256, 256, LDA, sB, tid);
        __syncthreads();

        // fc1 chunk (nt-outer, A hoisted to registers) + GELU -> sH
        v16h aReg[8];
#pragma unroll
        for (int k = 0; k < 8; ++k)
            aReg[k] = load_frag_a(sA + mt * 16 * LDA, LDA, k * 32, lane);
#pragma unroll
        for (int nt = 0; nt < 16; ++nt) {
            v8f hc = {};
#pragma unroll
            for (int k = 0; k < 8; ++k) {
                v16h bb = load_frag_bt(sB + nt * 16 * LDA, LDA, k * 32, lane);
                hc = wmma_f16(aReg[k], bb, hc);
            }
            const int col = c4 * 256 + nt * 16 + nl;
            const float bias = fc1b[col];
#pragma unroll
            for (int r = 0; r < 8; ++r)
                sH[(mt * 16 + mo + r) * LDA + nt * 16 + nl] = (_Float16)gelu_exact(hc[r] + bias);
        }
        __syncthreads();

        stage_bh(fc2t + c4 * 256, 1024, 256, 256, LDA, sB, tid);
        if (c4 < 3)  // warm GL2 with next chunk's fc1 weights
            __builtin_prefetch(fc1t + (size_t)((c4 + 1) * 256) * 256 + (size_t)tid * 256, 0, 1);
        __syncthreads();

        // fc2 partial (k-outer, single hoisted A per step)
        for (int k = 0; k < 8; ++k) {
            v16h a = load_frag_a(sH + mt * 16 * LDA, LDA, k * 32, lane);
#pragma unroll
            for (int nt = 0; nt < 16; ++nt) {
                v16h bb = load_frag_bt(sB + nt * 16 * LDA, LDA, k * 32, lane);
                acc[nt] = wmma_f16(a, bb, acc[nt]);
            }
        }
        __syncthreads();
    }

    // ---- epilogue: out = x1 + gamma2*(acc + fc2_b) ----
#pragma unroll
    for (int nt = 0; nt < 16; ++nt) {
        const int col = nt * 16 + nl;
        const float bias = fc2b[col], g = gamma2[col];
#pragma unroll
        for (int r = 0; r < 8; ++r) {
            const size_t gi = (T0 + mt * 16 + mo + r) * 256 + col;
            out[gi] = x1[gi] + g * (acc[nt][r] + bias);
        }
    }
}

// ---------------- host side ---------------------------------------------------
extern "C" void kernel_launch(void* const* d_in, const int* in_sizes, int n_in,
                              void* d_out, int out_size, void* d_ws, size_t ws_size,
                              hipStream_t stream) {
    const float* x        = (const float*)d_in[0];
    const float* norm1_s  = (const float*)d_in[1];
    const float* norm1_b  = (const float*)d_in[2];
    const float* gamma1   = (const float*)d_in[3];
    const float* aw1      = (const float*)d_in[4];
    const float* ab1      = (const float*)d_in[5];
    const float* aw2      = (const float*)d_in[6];
    const float* ab2      = (const float*)d_in[7];
    const float* wbank    = (const float*)d_in[8];
    const float* pre_w    = (const float*)d_in[9];
    const float* pre_b    = (const float*)d_in[10];
    const float* post_w   = (const float*)d_in[11];
    const float* post_b   = (const float*)d_in[12];
    const float* norm2_s  = (const float*)d_in[13];
    const float* norm2_b  = (const float*)d_in[14];
    const float* gamma2   = (const float*)d_in[15];
    const float* fc1_w    = (const float*)d_in[16];
    const float* fc1_b    = (const float*)d_in[17];
    const float* fc2_w    = (const float*)d_in[18];
    const float* fc2_b    = (const float*)d_in[19];
    const int*   rel_idx  = (const int*)d_in[20];
    float* out = (float*)d_out;

    char* ws = (char*)d_ws;
    _Float16* w1t   = (_Float16*)(ws + 0);        //  64x256 f16
    _Float16* w2t   = (_Float16*)(ws + 32768);    //  64x64  f16
    _Float16* pret  = (_Float16*)(ws + 40960);    // 256x256 f16
    _Float16* postt = (_Float16*)(ws + 172032);   // 256x256 f16
    _Float16* fc1t  = (_Float16*)(ws + 303104);   // 1024x256 f16
    _Float16* fc2t  = (_Float16*)(ws + 827392);   // 256x1024 f16
    float*    wbg   = (float*)   (ws + 1351680);  // 8x49x49 f32
    float*    x1    = (float*)   (ws + 1428736);  // B*56*56*256 f32 intermediate

    const int Bimg    = in_sizes[0] / (56 * 56 * 256);
    const int nWin    = Bimg * 64;
    const int nTokens = in_sizes[0] / 256;
    const int nTiles  = nTokens / 128;

    const int SMEM1 = 288768;  // win_kernel dynamic LDS
    const int SMEM2 = 270336;  // mlp_kernel dynamic LDS
    hipFuncSetAttribute((const void*)win_kernel, hipFuncAttributeMaxDynamicSharedMemorySize, SMEM1);
    hipFuncSetAttribute((const void*)mlp_kernel, hipFuncAttributeMaxDynamicSharedMemorySize, SMEM2);

    prep_kernel<<<2716, 256, 0, stream>>>(aw1, aw2, pre_w, post_w, fc1_w, fc2_w, wbank, rel_idx,
                                          w1t, w2t, pret, postt, fc1t, fc2t, wbg);

    win_kernel<<<nWin, 256, SMEM1, stream>>>(x, norm1_s, norm1_b, gamma1, ab1, ab2,
                                             pre_b, post_b, w1t, w2t, pret, postt, wbg, x1);

    mlp_kernel<<<nTiles, 256, SMEM2, stream>>>(x1, norm2_s, norm2_b, gamma2, fc1_b, fc2_b,
                                               fc1t, fc2t, out);
}
